// MLPModel_70703751626902
// MI455X (gfx1250) — matrix operand order repaired and checked by reading, not compiled, not run on hardware
//
#include <hip/hip_runtime.h>

#define N_ROBOTS 32
#define SEQ      12
#define LOBS     52
#define HID      256
#define LOUT     2
#define BATCH    1024
#define KIN      (SEQ * LOBS)   /* 624 */
#define KIN_PAD  640            /* 20 * 32 */
#define TILE_M   16
#define OUTD     (SEQ * LOUT)   /* 24 */

typedef __attribute__((ext_vector_type(16))) __bf16 v16bf;
typedef __attribute__((ext_vector_type(8)))  __bf16 v8bf;
typedef __attribute__((ext_vector_type(8)))  float  v8f;

// ---------------- bucketing kernels (group batch rows by robot id) -----------

__global__ void k_zero(int* ws) {
    int t = threadIdx.x;
    if (t < 96) ws[t] = 0;   // counts[32], offsets[32], cursor[32]
}

__global__ void k_count(const int* __restrict__ ids, int* __restrict__ cnt) {
    int b = blockIdx.x * blockDim.x + threadIdx.x;
    if (b < BATCH) atomicAdd(&cnt[ids[b]], 1);
}

__global__ void k_scan(const int* __restrict__ cnt, int* __restrict__ off) {
    if (threadIdx.x == 0) {
        int a = 0;
        for (int r = 0; r < N_ROBOTS; ++r) { off[r] = a; a += cnt[r]; }
    }
}

__global__ void k_scatter(const int* __restrict__ ids, const int* __restrict__ off,
                          int* __restrict__ cur, int* __restrict__ order) {
    int b = blockIdx.x * blockDim.x + threadIdx.x;
    if (b < BATCH) {
        int r = ids[b];
        int p = atomicAdd(&cur[r], 1);
        order[off[r] + p] = b;
    }
}

// ---------------- WMMA fragment loaders --------------------------------------

// 16-bit A fragment, 16x32 (MxK): lane<16 holds m=lane, K = k0..k0+7 and k0+16..k0+23
// (k0 = kk for lanes 0-15, kk+8 for lanes 16-31), 2 bf16 per VGPR.
__device__ __forceinline__ v16bf load_a(const __bf16* __restrict__ As, int lda,
                                        int m, int k0, int k1) {
    v8bf lo = *(const v8bf*)(As + m * lda + k0);
    v8bf hi = *(const v8bf*)(As + m * lda + k1);
    v16bf a;
#pragma unroll
    for (int i = 0; i < 8; ++i) { a[i] = lo[i]; a[8 + i] = hi[i]; }
    return a;
}

// Row load of B (not the WMMA B layout, where lanes index n): lane l holds row k = kk + l, n = nn..nn+15.
// Source is fp32 row-major [K][ldb]; convert to bf16 in-register.
__device__ __forceinline__ v16bf load_b(const float* __restrict__ Bg, int ldb,
                                        int krow, int nn) {
    const float4* q = (const float4*)(Bg + (size_t)krow * ldb + nn);
    v16bf b;
#pragma unroll
    for (int i = 0; i < 4; ++i) {
        float4 f = q[i];
        b[4 * i + 0] = (__bf16)f.x;
        b[4 * i + 1] = (__bf16)f.y;
        b[4 * i + 2] = (__bf16)f.z;
        b[4 * i + 3] = (__bf16)f.w;
    }
    return b;
}

// One GEMM phase: X[16,256] = relu(A[16,K] * B[K,256] + bias), X written as bf16 to LDS.
// Each wave covers 32 columns (2 N-tiles of 16). biasLDS: per-(m,n) [16*HID] f32, or null.
// biasG: per-n f32 (broadcast over m), or null. kmax clamps B row index (padding-safe).
__device__ __forceinline__ void gemm_phase(const __bf16* __restrict__ As, int lda,
                                           int kiters, int kmax,
                                           const float* __restrict__ Bg, int ldb,
                                           const float* __restrict__ biasLDS,
                                           const float* __restrict__ biasG,
                                           __bf16* __restrict__ Xout) {
    const int lane = threadIdx.x & 31;
    const int wave = threadIdx.x >> 5;
    const int hi   = lane >> 4;
    const int ln   = lane & 15;
#pragma unroll
    for (int nt = 0; nt < 2; ++nt) {
        const int nn = wave * 32 + nt * 16;
        v8f acc;
        if (biasLDS) {
#pragma unroll
            for (int v = 0; v < 8; ++v) acc[v] = biasLDS[(v + 8 * hi) * HID + nn + ln];
        } else {
            float bv = biasG[nn + ln];
#pragma unroll
            for (int v = 0; v < 8; ++v) acc[v] = bv;
        }
        const int kend = kiters * 32;
        for (int kk = 0; kk < kend; kk += 32) {
            const int k0 = kk + hi * 8;
            const int k1 = kk + 16 + hi * 8;
            v16bf a = load_a(As, lda, ln, k0, k1);
            v16bf b;
#pragma unroll
            for (int i = 0; i < 8; ++i) {
                int ka = k0 + i; if (ka > kmax) ka = kmax;
                int kb = k1 + i; if (kb > kmax) kb = kmax;
                b[i]     = (__bf16)Bg[(size_t)ka * ldb + nn + ln];
                b[8 + i] = (__bf16)Bg[(size_t)kb * ldb + nn + ln];
            }
            if (kk + 32 < kend) {
                int pk = kk + 32 + lane; if (pk > kmax) pk = kmax;
                __builtin_prefetch(Bg + (size_t)pk * ldb + nn, 0, 1);
            }
            acc = __builtin_amdgcn_wmma_f32_16x16x32_bf16(
                false, a, false, b, (short)0, acc, false, false);
        }
        // epilogue: relu -> bf16 -> LDS (C/D layout: m = v + 8*hi, n = nn + ln)
#pragma unroll
        for (int v = 0; v < 8; ++v) {
            float x = acc[v];
            x = x > 0.f ? x : 0.f;
            Xout[(v + 8 * hi) * HID + nn + ln] = (__bf16)x;
        }
    }
}

// ---------------- main fused-MLP kernel --------------------------------------

__global__ __launch_bounds__(256, 2)
void k_mlp(const float* __restrict__ obs, const unsigned char* __restrict__ mask,
           const float* __restrict__ Wi, const float* __restrict__ bi,
           const float* __restrict__ W1, const float* __restrict__ b1,
           const float* __restrict__ W2, const float* __restrict__ b2,
           const float* __restrict__ W3, const float* __restrict__ b3,
           const float* __restrict__ Wo, const float* __restrict__ bo,
           const int* __restrict__ cnt, const int* __restrict__ off,
           const int* __restrict__ order, float* __restrict__ out) {
    const int r  = blockIdx.x;          // robot
    const int t  = blockIdx.y;          // 16-row tile within this robot's bucket
    const int nb = cnt[r];
    if (t * TILE_M >= nb) return;       // uniform across block: safe for WMMA EXEC

    __shared__ __bf16 obsbuf[TILE_M * KIN_PAD];   // 20480 B, masked obs (bf16)
    __shared__ __bf16 xA[TILE_M * HID];           //  8192 B
    __shared__ __bf16 xB[TILE_M * HID];           //  8192 B
    __shared__ float  biasbuf[TILE_M * HID];      // 16384 B, per-sample input bias
    __shared__ float  scalebuf[TILE_M * SEQ];     // (1 - mask) per (row, limb)
    __shared__ int    rows[TILE_M];

    const int tid = threadIdx.x;

    if (tid < TILE_M) {
        int idx = t * TILE_M + tid;
        rows[tid] = (idx < nb) ? order[off[r] + idx] : -1;
    }
    __syncthreads();

    // stage masked obs (exact: mask is linear in per-limb embedding)
    for (int e = tid; e < TILE_M * KIN; e += 256) {
        int i = e / KIN, c = e % KIN, s = c / LOBS;
        int bidx = rows[i];
        float v = 0.f;
        if (bidx >= 0 && !mask[bidx * SEQ + s]) v = obs[(size_t)bidx * KIN + c];
        obsbuf[i * KIN_PAD + c] = (__bf16)v;
    }
    // zero the K padding (624..639) so clamped B rows contribute nothing
    {
        int i = tid >> 4, c = KIN + (tid & 15);
        obsbuf[i * KIN_PAD + c] = (__bf16)0.f;
    }
    if (tid < TILE_M * SEQ) {
        int i = tid / SEQ, s = tid % SEQ;
        int bidx = rows[i];
        scalebuf[tid] = (bidx >= 0 && !mask[bidx * SEQ + s]) ? 1.f : 0.f;
    }
    __syncthreads();

    // per-sample input bias: bias[i][n] = sum_s (1-m[i][s]) * bi[r][s][n]
    {
        const int n = tid;
        float biv[SEQ];
#pragma unroll
        for (int s = 0; s < SEQ; ++s) biv[s] = bi[((size_t)r * SEQ + s) * HID + n];
        for (int i = 0; i < TILE_M; ++i) {
            float sum = 0.f;
#pragma unroll
            for (int s = 0; s < SEQ; ++s) sum += scalebuf[i * SEQ + s] * biv[s];
            biasbuf[i * HID + n] = sum;
        }
    }
    __syncthreads();

    // input hyper-layer: [16,624] x [624,256] (K padded to 640, B rows clamped)
    gemm_phase(obsbuf, KIN_PAD, KIN_PAD / 32, KIN - 1,
               Wi + (size_t)r * KIN * HID, HID, biasbuf, nullptr, xA);
    __syncthreads();
    // three per-robot hidden layers: [16,256] x [256,256]
    gemm_phase(xA, HID, HID / 32, HID - 1,
               W1 + (size_t)r * HID * HID, HID, nullptr, b1 + (size_t)r * HID, xB);
    __syncthreads();
    gemm_phase(xB, HID, HID / 32, HID - 1,
               W2 + (size_t)r * HID * HID, HID, nullptr, b2 + (size_t)r * HID, xA);
    __syncthreads();
    gemm_phase(xA, HID, HID / 32, HID - 1,
               W3 + (size_t)r * HID * HID, HID, nullptr, b3 + (size_t)r * HID, xB);
    __syncthreads();

    // tiny output hyper-layer: out[b,s,d] = x . Wo[r,s,:,d] + bo[r,s,d]  (24 cols)
    for (int e = tid; e < TILE_M * OUTD; e += 256) {
        int i = e / OUTD, j = e % OUTD;
        int bidx = rows[i];
        if (bidx >= 0) {
            int s = j >> 1, d = j & 1;
            const float* wo = Wo + ((size_t)r * SEQ + s) * HID * LOUT + d;
            const __bf16* xr = xB + i * HID;
            float sum = bo[((size_t)r * SEQ + s) * LOUT + d];
            for (int h = 0; h < HID; ++h)
                sum += (float)xr[h] * wo[(size_t)h * LOUT];
            out[(size_t)bidx * OUTD + j] = sum;
        }
    }
}

// ---------------- launch -----------------------------------------------------

extern "C" void kernel_launch(void* const* d_in, const int* in_sizes, int n_in,
                              void* d_out, int out_size, void* d_ws, size_t ws_size,
                              hipStream_t stream) {
    const float*         obs  = (const float*)d_in[0];
    const unsigned char* mask = (const unsigned char*)d_in[1];
    const int*           ids  = (const int*)d_in[2];
    const float* Wi = (const float*)d_in[3];
    const float* bi = (const float*)d_in[4];
    const float* W1 = (const float*)d_in[5];
    const float* b1 = (const float*)d_in[6];
    const float* W2 = (const float*)d_in[7];
    const float* b2 = (const float*)d_in[8];
    const float* W3 = (const float*)d_in[9];
    const float* b3 = (const float*)d_in[10];
    const float* Wo = (const float*)d_in[11];
    const float* bo = (const float*)d_in[12];

    int* ws    = (int*)d_ws;
    int* cnt   = ws;        // 32
    int* off   = ws + 32;   // 32
    int* cur   = ws + 64;   // 32
    int* order = ws + 96;   // 1024

    k_zero   <<<1, 128, 0, stream>>>(ws);
    k_count  <<<BATCH / 256, 256, 0, stream>>>(ids, cnt);
    k_scan   <<<1, 32, 0, stream>>>(cnt, off);
    k_scatter<<<BATCH / 256, 256, 0, stream>>>(ids, off, cur, order);

    dim3 grid(N_ROBOTS, BATCH / TILE_M);   // 32 x 64, blocks past bucket end exit early
    k_mlp<<<grid, 256, 0, stream>>>(obs, mask, Wi, bi, W1, b1, W2, b2, W3, b3,
                                    Wo, bo, cnt, off, order, (float*)d_out);
}
